// DTIModel_20761871909536
// MI455X (gfx1250) — matrix operand order repaired and checked by reading, not compiled, not run on hardware
//
#include <hip/hip_runtime.h>
#include <hip/hip_bf16.h>

#define Hd    128
#define NP    512
#define ND    96
#define NBATCH 8
#define DEPTH 3

typedef __bf16 bf16t;
typedef __attribute__((ext_vector_type(16))) __bf16 v16bf;
typedef __attribute__((ext_vector_type(8)))  float  v8f;

__device__ __forceinline__ float lrelu(float x, float s) { return x >= 0.f ? x : s * x; }

__device__ __forceinline__ v8f wmma_bf16(v16bf a, v16bf b, v8f c) {
  // D = A(16x32 bf16) x B(32x16 bf16) + C(16x16 f32)
  return __builtin_amdgcn_wmma_f32_16x16x32_bf16(false, a, false, b, (short)0, c, false, false);
}

// ---------------------------------------------------------------------------
// Weight prep: f32 row-major W[k][n] (128x128) -> bf16 in WMMA B-fragment order.
// Fragment (nb,kb): lane holds 16 contiguous bf16 = column n=nb*16+(lane&15),
// rows k = kb*32 + (e>>3)*16 + (lane>>4)*8 + (e&7)  (ISA 7.12.2 B-matrix striping).
// ---------------------------------------------------------------------------
__global__ void prep_weight_k(const float* __restrict__ W, bf16t* __restrict__ out) {
  int idx  = blockIdx.x * 256 + threadIdx.x;        // 16384 elements
  int e    = idx & 15;
  int lane = (idx >> 4) & 31;
  int blk  = idx >> 9;                               // nb*4 + kb
  int kb   = blk & 3, nb = blk >> 2;
  int k = kb * 32 + (e >> 3) * 16 + (lane >> 4) * 8 + (e & 7);
  int n = nb * 16 + (lane & 15);
  out[idx] = (bf16t)W[k * Hd + n];
}

__device__ __forceinline__ v16bf load_bfrag(const bf16t* wp, int nb, int kb, int lane) {
  return *(const v16bf*)(wp + ((((nb << 2) + kb) * 32 + lane) << 4));   // 32B contiguous
}

// A fragment (16x32, MxK) from f32 LDS rows: row = rowBase+(lane&15),
// K layout per ISA: g<4 -> k=2g(+8 if lane>=16); g>=4 -> k=16+2(g-4)(+8 if lane>=16)
__device__ __forceinline__ v16bf load_afrag(const float* xs, int stride,
                                            int rowBase, int k0, int lane) {
  int row = rowBase + (lane & 15);
  const float* p = xs + row * stride + k0 + ((lane >> 4) << 3);
  v16bf a;
#pragma unroll
  for (int g = 0; g < 8; ++g) {
    int kk = (g < 4) ? (g << 1) : (16 + ((g - 4) << 1));
    a[2 * g]     = (bf16t)p[kk];
    a[2 * g + 1] = (bf16t)p[kk + 1];
  }
  return a;
}

#define PSTR 130
__device__ __forceinline__ v16bf load_afrag_fma(const float* m, const float* t,
                                                const float* alpha,
                                                int rowBase, int k0, int lane) {
  int row = rowBase + (lane & 15);
  float al = alpha[row];
  const float* pm = m + row * PSTR + k0 + ((lane >> 4) << 3);
  const float* pt = t + row * PSTR + k0 + ((lane >> 4) << 3);
  v16bf a;
#pragma unroll
  for (int g = 0; g < 8; ++g) {
    int kk = (g < 4) ? (g << 1) : (16 + ((g - 4) << 1));
    a[2 * g]     = (bf16t)(pm[kk]     + al * pt[kk]);
    a[2 * g + 1] = (bf16t)(pm[kk + 1] + al * pt[kk + 1]);
  }
  return a;
}

// ---------------------------------------------------------------------------
// Y[R,128] = act(X[R,128] @ W + bias) * mask ; 64 rows per WG, 4 waves x 16 rows.
// ---------------------------------------------------------------------------
#define LSTR 132
__global__ void __launch_bounds__(128) linear128_k(
    const float* __restrict__ X, const bf16t* __restrict__ Wp,
    const float* __restrict__ bias, const float* __restrict__ mask,
    float* __restrict__ Y, int actMode, float slope) {
  __shared__ float xs[64 * LSTR];
  int tid = threadIdx.x;
  int row0 = blockIdx.x * 64;
  for (int i = tid; i < 64 * 32; i += 128) {
    int r = i >> 5, c4 = (i & 31) << 2;
    float4 v = *(const float4*)(X + (size_t)(row0 + r) * Hd + c4);
    *(float4*)(xs + r * LSTR + c4) = v;
  }
  __syncthreads();
  int wid = tid >> 5, lane = tid & 31;
  int rowBase = wid * 16;
  v16bf a0 = load_afrag(xs, LSTR, rowBase, 0,  lane);
  v16bf a1 = load_afrag(xs, LSTR, rowBase, 32, lane);
  v16bf a2 = load_afrag(xs, LSTR, rowBase, 64, lane);
  v16bf a3 = load_afrag(xs, LSTR, rowBase, 96, lane);
#pragma unroll
  for (int nb = 0; nb < 8; ++nb) {
    v8f c = {};
    c = wmma_bf16(a0, load_bfrag(Wp, nb, 0, lane), c);
    c = wmma_bf16(a1, load_bfrag(Wp, nb, 1, lane), c);
    c = wmma_bf16(a2, load_bfrag(Wp, nb, 2, lane), c);
    c = wmma_bf16(a3, load_bfrag(Wp, nb, 3, lane), c);
    int col = nb * 16 + (lane & 15);
    int rlo = (lane >> 4) << 3;
    float bia = bias ? bias[col] : 0.f;
#pragma unroll
    for (int r = 0; r < 8; ++r) {
      int row = row0 + rowBase + rlo + r;
      float v = c[r] + bia;
      if (actMode == 1) v = lrelu(v, slope);
      if (mask) v *= mask[row];
      Y[(size_t)row * Hd + col] = v;
    }
  }
}

// ---------------------------------------------------------------------------
// Fused pair kernel: builds 32 mij rows in LDS, runs 3 attention iterations
// (GEMM + alpha + rank-1 update) and the W1a/W1b head, emits per-WG column sums.
// 4 waves: wave owns rows ((wid>>1)*16 .. +15) x col-tiles ((wid&1)*4 .. +3).
// ---------------------------------------------------------------------------
#define PROWS 32
#define BLKS_PER_B (NP * ND / PROWS)   // 1536
__global__ void __launch_bounds__(128) pair_kernel(
    const float* __restrict__ pd, const float* __restrict__ dd,
    const bf16t* __restrict__ WvsP, const float* __restrict__ Wvs_b,
    const float* __restrict__ attW, const float* __restrict__ attB,
    const bf16t* __restrict__ W1aP, const float* __restrict__ W1a_b,
    const bf16t* __restrict__ W1bP, const float* __restrict__ W1b_b,
    float* __restrict__ partials) {
  __shared__ float m[PROWS * PSTR];
  __shared__ float t[PROWS * PSTR];
  __shared__ float alpha[PROWS];
  __shared__ float scr[128];
  __shared__ float ps[4 * 128];
  int tid = threadIdx.x;
  int wg = blockIdx.x;
  int b = wg / BLKS_PER_B;
  int blk = wg - b * BLKS_PER_B;
  int prow0 = blk * PROWS;

  // m = lrelu(pd_row + dd_row, 0.1)   (Wv_b was folded into dd)
  for (int i = tid; i < PROWS * Hd; i += 128) {
    int r = i >> 7, h = i & 127;
    int rg = prow0 + r;
    int p = rg / ND, d = rg - p * ND;
    float v = pd[((size_t)b * NP + p) * Hd + h] + dd[((size_t)b * ND + d) * Hd + h];
    m[r * PSTR + h] = lrelu(v, 0.1f);
  }
  __syncthreads();

  int wid = tid >> 5, lane = tid & 31;
  int rowBase = (wid >> 1) * 16;
  int nb0 = (wid & 1) * 4;

  for (int it = 0; it < DEPTH; ++it) {
    const bf16t* Wp = WvsP + it * 16384;
    const float* wb = Wvs_b + it * Hd;
    if (it + 1 < DEPTH) __builtin_prefetch(WvsP + (it + 1) * 16384, 0, 1);
    v16bf a0 = load_afrag(m, PSTR, rowBase, 0,  lane);
    v16bf a1 = load_afrag(m, PSTR, rowBase, 32, lane);
    v16bf a2 = load_afrag(m, PSTR, rowBase, 64, lane);
    v16bf a3 = load_afrag(m, PSTR, rowBase, 96, lane);
#pragma unroll
    for (int j = 0; j < 4; ++j) {
      int nb = nb0 + j;
      v8f c = {};
      c = wmma_bf16(a0, load_bfrag(Wp, nb, 0, lane), c);
      c = wmma_bf16(a1, load_bfrag(Wp, nb, 1, lane), c);
      c = wmma_bf16(a2, load_bfrag(Wp, nb, 2, lane), c);
      c = wmma_bf16(a3, load_bfrag(Wp, nb, 3, lane), c);
      int col = nb * 16 + (lane & 15);
      int rlo = (lane >> 4) << 3;
      float bia = wb[col];
#pragma unroll
      for (int r = 0; r < 8; ++r)
        t[(rowBase + rlo + r) * PSTR + col] = c[r] + bia;
    }
    __syncthreads();
    // alpha[r] = t_r . att[it] + att_b[it]   (4 threads per row, fixed-order combine)
    {
      int r = tid >> 2, q = tid & 3;
      const float* tr = t + r * PSTR + q * 32;
      const float* aw = attW + it * Hd + q * 32;
      float s = 0.f;
#pragma unroll
      for (int k2 = 0; k2 < 32; ++k2) s += tr[k2] * aw[k2];
      scr[tid] = s;
    }
    __syncthreads();
    if (tid < PROWS)
      alpha[tid] = scr[tid * 4] + scr[tid * 4 + 1] + scr[tid * 4 + 2] + scr[tid * 4 + 3] + attB[it];
    __syncthreads();
    // m += alpha * t
    for (int i = tid; i < PROWS * Hd; i += 128) {
      int r = i >> 7, h = i & 127;
      m[r * PSTR + h] += alpha[r] * t[r * PSTR + h];
    }
    __syncthreads();
  }

  // final = m + alpha_last * t ; h1 = relu(final @ W1a + b)  (accumulate all tiles,
  // then barrier before overwriting m, since partner wave reads all columns)
  v16bf f0 = load_afrag_fma(m, t, alpha, rowBase, 0,  lane);
  v16bf f1 = load_afrag_fma(m, t, alpha, rowBase, 32, lane);
  v16bf f2 = load_afrag_fma(m, t, alpha, rowBase, 64, lane);
  v16bf f3 = load_afrag_fma(m, t, alpha, rowBase, 96, lane);
  v8f acc[4];
#pragma unroll
  for (int j = 0; j < 4; ++j) {
    int nb = nb0 + j;
    v8f c = {};
    c = wmma_bf16(f0, load_bfrag(W1aP, nb, 0, lane), c);
    c = wmma_bf16(f1, load_bfrag(W1aP, nb, 1, lane), c);
    c = wmma_bf16(f2, load_bfrag(W1aP, nb, 2, lane), c);
    c = wmma_bf16(f3, load_bfrag(W1aP, nb, 3, lane), c);
    acc[j] = c;
  }
  __syncthreads();
#pragma unroll
  for (int j = 0; j < 4; ++j) {
    int nb = nb0 + j;
    int col = nb * 16 + (lane & 15);
    int rlo = (lane >> 4) << 3;
    float bia = W1a_b[col];
#pragma unroll
    for (int r = 0; r < 8; ++r) {
      float v = acc[j][r] + bia;
      m[(rowBase + rlo + r) * PSTR + col] = v > 0.f ? v : 0.f;
    }
  }
  __syncthreads();

  // h2 = h1 @ W1b + b ; per-WG column partial sums (deterministic tree)
  v16bf g0 = load_afrag(m, PSTR, rowBase, 0,  lane);
  v16bf g1f = load_afrag(m, PSTR, rowBase, 32, lane);
  v16bf g2f = load_afrag(m, PSTR, rowBase, 64, lane);
  v16bf g3f = load_afrag(m, PSTR, rowBase, 96, lane);
#pragma unroll
  for (int j = 0; j < 4; ++j) {
    int nb = nb0 + j;
    v8f c = {};
    c = wmma_bf16(g0,  load_bfrag(W1bP, nb, 0, lane), c);
    c = wmma_bf16(g1f, load_bfrag(W1bP, nb, 1, lane), c);
    c = wmma_bf16(g2f, load_bfrag(W1bP, nb, 2, lane), c);
    c = wmma_bf16(g3f, load_bfrag(W1bP, nb, 3, lane), c);
    int col = nb * 16 + (lane & 15);
    float s = 8.f * W1b_b[col];   // 8 rows per partial include the bias
#pragma unroll
    for (int r = 0; r < 8; ++r) s += c[r];
    ps[(((wid >> 1) << 1) + (lane >> 4)) * 128 + col] = s;
  }
  __syncthreads();
  if (tid < 128) {
    float s = ps[tid] + ps[128 + tid] + ps[256 + tid] + ps[384 + tid];
    partials[(size_t)wg * 128 + tid] = s;
  }
}

// ---------------------------------------------------------------------------
__global__ void sum_rows_k(const float* __restrict__ X, float* __restrict__ out, int N) {
  int b = blockIdx.x, h = threadIdx.x;
  float s = 0.f;
  for (int r = 0; r < N; ++r) s += X[((size_t)(b * N + r)) * Hd + h];
  out[b * Hd + h] = s;
}

__global__ void reduce_partials_k(const float* __restrict__ part, float* __restrict__ g1) {
  int b = blockIdx.x, h = threadIdx.x;
  float s = 0.f;
  for (int k = 0; k < BLKS_PER_B; ++k) s += part[((size_t)(b * BLKS_PER_B + k)) * 128 + h];
  g1[b * 128 + h] = s;
}

// ---------------------------------------------------------------------------
// Tail MLP on [8, ...] rows — trivially small, scalar, one block.
// ---------------------------------------------------------------------------
__global__ void __launch_bounds__(256) tail_k(
    const float* __restrict__ u_d, const float* __restrict__ u_p, const float* __restrict__ g1,
    const float* __restrict__ WuW, const float* __restrict__ Wub,
    const float* __restrict__ W2aW, const float* __restrict__ W2ab,
    const float* __restrict__ W2bW, const float* __restrict__ W2bb,
    const float* __restrict__ W3W, const float* __restrict__ W3b,
    const float* __restrict__ Wu1aW, const float* __restrict__ Wu1ab,
    const float* __restrict__ Wu1bW, const float* __restrict__ Wu1bb,
    const float* __restrict__ W5W, const float* __restrict__ W5b,
    float* __restrict__ out) {
  __shared__ float cat[256], t256[256], mu_s[128], g2[128], g3[128], v1[128], v2[128];
  int tid = threadIdx.x;
  for (int b = 0; b < NBATCH; ++b) {
    if (tid < 128) cat[tid] = u_d[b * 128 + tid];
    else           cat[tid] = u_p[b * 128 + tid - 128];
    __syncthreads();
    if (tid < 128) {
      float s = Wub[tid];
      for (int k = 0; k < 256; ++k) s += cat[k] * WuW[k * 128 + tid];
      mu_s[tid] = lrelu(s, 0.01f);
    }
    {
      float s = W2ab[tid];
      for (int k = 0; k < 128; ++k) s += g1[b * 128 + k] * W2aW[k * 256 + tid];
      t256[tid] = lrelu(s, 0.1f);
    }
    __syncthreads();
    if (tid < 128) {
      float s = W2bb[tid];
      for (int k = 0; k < 256; ++k) s += t256[k] * W2bW[k * 128 + tid];
      g2[tid] = s;
    }
    __syncthreads();
    if (tid < 128) {
      float s = W3b[tid];
      for (int k = 0; k < 128; ++k) s += g2[k] * W3W[k * 128 + tid];
      g3[tid] = lrelu(s, 0.1f);
    }
    __syncthreads();
    if (tid < 128) cat[tid] = mu_s[tid];
    else           cat[tid] = g3[tid - 128];
    __syncthreads();
    if (tid < 128) {
      float s = Wu1ab[tid];
      for (int k = 0; k < 256; ++k) s += cat[k] * Wu1aW[k * 128 + tid];
      v1[tid] = lrelu(s, 0.1f);
    }
    __syncthreads();
    if (tid < 128) {
      float s = Wu1bb[tid];
      for (int k = 0; k < 128; ++k) s += v1[k] * Wu1bW[k * 128 + tid];
      v2[tid] = s;
    }
    __syncthreads();
    if (tid == 0) {
      float s = W5b[0];
      for (int k = 0; k < 128; ++k) s += v2[k] * W5W[k];
      out[b] = s;
    }
    __syncthreads();
  }
}

// ---------------------------------------------------------------------------
extern "C" void kernel_launch(void* const* d_in, const int* in_sizes, int n_in,
                              void* d_out, int out_size, void* d_ws, size_t ws_size,
                              hipStream_t stream) {
  (void)in_sizes; (void)n_in; (void)out_size; (void)ws_size;
  const float* protein = (const float*)d_in[0];
  const float* drug    = (const float*)d_in[1];
  const float* pu_mask = (const float*)d_in[2];
  const float* du_mask = (const float*)d_in[3];
  const float* d_W  = (const float*)d_in[4];   const float* d_b  = (const float*)d_in[5];
  const float* p_W  = (const float*)d_in[6];   const float* p_b  = (const float*)d_in[7];
  const float* Wv_p = (const float*)d_in[8];   const float* Wv_d = (const float*)d_in[9];
  const float* Wv_b = (const float*)d_in[10];
  const float* att_W = (const float*)d_in[11]; const float* att_b = (const float*)d_in[12];
  const float* Wvs_W = (const float*)d_in[13]; const float* Wvs_b = (const float*)d_in[14];
  const float* Wu_W  = (const float*)d_in[15]; const float* Wu_b  = (const float*)d_in[16];
  const float* W1a_W = (const float*)d_in[17]; const float* W1a_b = (const float*)d_in[18];
  const float* W1b_W = (const float*)d_in[19]; const float* W1b_b = (const float*)d_in[20];
  const float* W2a_W = (const float*)d_in[21]; const float* W2a_b = (const float*)d_in[22];
  const float* W2b_W = (const float*)d_in[23]; const float* W2b_b = (const float*)d_in[24];
  const float* W3_W  = (const float*)d_in[25]; const float* W3_b  = (const float*)d_in[26];
  const float* Wu1a_W = (const float*)d_in[27]; const float* Wu1a_b = (const float*)d_in[28];
  const float* Wu1b_W = (const float*)d_in[29]; const float* Wu1b_b = (const float*)d_in[30];
  const float* W5_W  = (const float*)d_in[31]; const float* W5_b  = (const float*)d_in[32];
  float* out = (float*)d_out;

  char* ws = (char*)d_ws;
  size_t off = 0;
  auto alloc = [&](size_t bytes) -> void* {
    void* p = ws + off;
    off += (bytes + 255) & ~(size_t)255;
    return p;
  };
  float* prot_feat = (float*)alloc((size_t)NBATCH * NP * Hd * 4);
  float* drug_feat = (float*)alloc((size_t)NBATCH * ND * Hd * 4);
  float* pd        = (float*)alloc((size_t)NBATCH * NP * Hd * 4);
  float* dd        = (float*)alloc((size_t)NBATCH * ND * Hd * 4);
  float* u_d       = (float*)alloc(NBATCH * Hd * 4);
  float* u_p       = (float*)alloc(NBATCH * Hd * 4);
  float* g1        = (float*)alloc(NBATCH * Hd * 4);
  float* partials  = (float*)alloc((size_t)NBATCH * BLKS_PER_B * 128 * 4);
  bf16t* pw_d   = (bf16t*)alloc(16384 * 2);
  bf16t* pw_p   = (bf16t*)alloc(16384 * 2);
  bf16t* pw_wvp = (bf16t*)alloc(16384 * 2);
  bf16t* pw_wvd = (bf16t*)alloc(16384 * 2);
  bf16t* pw_wvs = (bf16t*)alloc((size_t)DEPTH * 16384 * 2);
  bf16t* pw_w1a = (bf16t*)alloc(16384 * 2);
  bf16t* pw_w1b = (bf16t*)alloc(16384 * 2);

  // 1. weight prep (bf16 fragment layout; all stay L2-resident afterwards)
  prep_weight_k<<<64, 256, 0, stream>>>(d_W,  pw_d);
  prep_weight_k<<<64, 256, 0, stream>>>(p_W,  pw_p);
  prep_weight_k<<<64, 256, 0, stream>>>(Wv_p, pw_wvp);
  prep_weight_k<<<64, 256, 0, stream>>>(Wv_d, pw_wvd);
  for (int i = 0; i < DEPTH; ++i)
    prep_weight_k<<<64, 256, 0, stream>>>(Wvs_W + (size_t)i * 16384, pw_wvs + (size_t)i * 16384);
  prep_weight_k<<<64, 256, 0, stream>>>(W1a_W, pw_w1a);
  prep_weight_k<<<64, 256, 0, stream>>>(W1b_W, pw_w1b);

  // 2. first-layer features (lrelu 0.1 then mask) and the Wv projections
  linear128_k<<<(NBATCH * ND) / 64, 128, 0, stream>>>(drug,    pw_d,   d_b,  du_mask, drug_feat, 1, 0.1f);
  linear128_k<<<(NBATCH * NP) / 64, 128, 0, stream>>>(protein, pw_p,   p_b,  pu_mask, prot_feat, 1, 0.1f);
  linear128_k<<<(NBATCH * NP) / 64, 128, 0, stream>>>(prot_feat, pw_wvp, nullptr, nullptr, pd, 0, 0.f);
  linear128_k<<<(NBATCH * ND) / 64, 128, 0, stream>>>(drug_feat, pw_wvd, Wv_b,    nullptr, dd, 0, 0.f);

  // 3. pooled features
  sum_rows_k<<<NBATCH, 128, 0, stream>>>(drug_feat, u_d, ND);
  sum_rows_k<<<NBATCH, 128, 0, stream>>>(prot_feat, u_p, NP);

  // 4. the fused heavy kernel: mij lives only in LDS
  pair_kernel<<<NBATCH * BLKS_PER_B, 128, 0, stream>>>(
      pd, dd, pw_wvs, Wvs_b, att_W, att_b, pw_w1a, W1a_b, pw_w1b, W1b_b, partials);

  // 5. deterministic reduction + tiny tail MLP
  reduce_partials_k<<<NBATCH, 128, 0, stream>>>(partials, g1);
  tail_k<<<1, 256, 0, stream>>>(u_d, u_p, g1,
                                Wu_W, Wu_b, W2a_W, W2a_b, W2b_W, W2b_b,
                                W3_W, W3_b, Wu1a_W, Wu1a_b, Wu1b_W, Wu1b_b,
                                W5_W, W5_b, out);
}
